// SimpleGrowthNCA_13580686590088
// MI455X (gfx1250) — matrix operand order repaired and checked by reading, not compiled, not run on hardware
//
#include <hip/hip_runtime.h>

typedef __attribute__((ext_vector_type(16))) __bf16 v16bf;
typedef __attribute__((ext_vector_type(8)))  float  v8f;

#define B_  16
#define C_  8
#define H_  512
#define W_  512
#define HW_ (H_ * W_)
#define NPIX_ (B_ * HW_)   // alive-mask element count (B*1*H*W)

// zero-padded load from one channel plane
__device__ __forceinline__ float ldz(const float* __restrict__ plane, int y, int x) {
    return ((unsigned)y < (unsigned)H_ && (unsigned)x < (unsigned)W_) ? plane[y * W_ + x] : 0.0f;
}

// ---------------- pass 1: global alive-ratio reduction (channel 3 only) ----------------
__global__ void __launch_bounds__(256) alive_count_kernel(const float* __restrict__ x,
                                                          unsigned* __restrict__ cnt) {
    __shared__ unsigned sred[256];
    const unsigned tid = threadIdx.x;
    unsigned gid = blockIdx.x * blockDim.x + tid;
    const unsigned stride = gridDim.x * blockDim.x;
    unsigned local = 0;
    for (unsigned i = gid; i < (unsigned)NPIX_; i += stride) {
        unsigned bI = i / (unsigned)HW_;
        unsigned p  = i % (unsigned)HW_;
        local += (x[(bI * C_ + 3) * HW_ + p] > 0.1f) ? 1u : 0u;
    }
    sred[tid] = local;
    __syncthreads();
    for (unsigned s = 128; s > 0; s >>= 1) {
        if (tid < s) sred[tid] += sred[tid + s];
        __syncthreads();
    }
    if (tid == 0) atomicAdd(cnt, sred[0]);
}

// ---------------- pass 2: fused perceive + MLP (bf16 WMMA) + boost + clamp ----------------
__global__ void __launch_bounds__(256) nca_fused_kernel(
    const float* __restrict__ x,
    const float* __restrict__ w1, const float* __restrict__ b1,
    const float* __restrict__ w2, const float* __restrict__ b2,
    const float* __restrict__ w3, const float* __restrict__ b3,
    const float* __restrict__ gr,
    const unsigned* __restrict__ aliveCnt,
    float* __restrict__ out)
{
    // per-wave private LDS region: 16 pixels x 33 floats (padded) for transposes,
    // reused (stride 9) for the epilogue dx scatter. 8 waves * 544 floats.
    __shared__ float sm[8 * 544];
    const int lane      = threadIdx.x & 31;
    const int waveInBlk = threadIdx.x >> 5;
    float* lds = sm + waveInBlk * 544;

    const int gw = blockIdx.x * 8 + waveInBlk;  // one wave per image row: 0 .. B*H-1
    const int b  = gw >> 9;                     // / H_
    const int y  = gw & (H_ - 1);

    const int n = lane & 15;                    // N index (output feature) in B/D fragments
    const int h = lane >> 4;                    // half-wave (K-range selector)
    const int m = lane & 15;                    // M index (pixel within tile) in A fragments

    const float g          = gr[0];             // growth_rate * GROWTH_BOOST(1.0)
    const float aliveRatio = (float)(*aliveCnt) * (1.0f / (float)NPIX_);
    const bool  applyBoost = aliveRatio < 0.2f;

    // ---- pack weight B-fragments once per wave (B layout: lanes0-15 K{0..7,16..23}, lanes16-31 K{8..15,24..31}, N=lane&15)
    v16bf B1a, B1b, B2a, B2b, B3f;
    const float bz1a = b1[n], bz1b = b1[n + 16];
    const float bz2a = b2[n], bz2b = b2[n + 16];
    const float bz3  = (n < 8) ? b3[n] : 0.0f;
#pragma unroll
    for (int e = 0; e < 16; ++e) {
        const int k = (e < 8) ? (h * 8 + e) : (16 + h * 8 + (e - 8));
        B1a[e] = (k < 24) ? (__bf16)w1[n * 24 + k]        : (__bf16)0.0f;
        B1b[e] = (k < 24) ? (__bf16)w1[(n + 16) * 24 + k] : (__bf16)0.0f;
        B2a[e] = (__bf16)w2[n * 32 + k];
        B2b[e] = (__bf16)w2[(n + 16) * 32 + k];
        B3f[e] = (n < 8) ? (__bf16)w3[n * 32 + k] : (__bf16)0.0f;
    }

    const float* xb = x + (size_t)b * C_ * HW_;
    const v8f z = {};

    for (int tx = 0; tx < W_ / 16; ++tx) {
        const int xc = tx * 16 + m;

        // prefetch next tile's center row into cache hierarchy
        if (tx + 1 < W_ / 16) {
#pragma unroll
            for (int c = 0; c < C_; ++c)
                __builtin_prefetch(&xb[c * HW_ + y * W_ + xc + 16], 0, 3);
        }

        // ---- perceive -> A1 (16-bit A layout 16x32: lanes0-15 K{0..7,16..23}, lanes16-31 K{8..15,24..31})
        v16bf A1;
        if (h == 0) {
#pragma unroll
            for (int c = 0; c < C_; ++c) {
                const float* pc = xb + c * HW_;
                A1[c] = (__bf16)pc[y * W_ + xc];                 // K = c       (identity)
                float sy = -ldz(pc, y - 1, xc - 1) - 2.0f * ldz(pc, y - 1, xc) - ldz(pc, y - 1, xc + 1)
                         +  ldz(pc, y + 1, xc - 1) + 2.0f * ldz(pc, y + 1, xc) + ldz(pc, y + 1, xc + 1);
                A1[8 + c] = (__bf16)sy;                          // K = 16 + c  (sobel_y)
            }
        } else {
#pragma unroll
            for (int c = 0; c < C_; ++c) {
                const float* pc = xb + c * HW_;
                float sx = -ldz(pc, y - 1, xc - 1) + ldz(pc, y - 1, xc + 1)
                         - 2.0f * ldz(pc, y, xc - 1) + 2.0f * ldz(pc, y, xc + 1)
                         -  ldz(pc, y + 1, xc - 1) + ldz(pc, y + 1, xc + 1);
                A1[c]     = (__bf16)sx;                          // K = 8 + c   (sobel_x)
                A1[8 + c] = (__bf16)0.0f;                        // K = 24 + c  (pad)
            }
        }

        // ---- layer 1: 24(->32 padded) -> 32
        v8f d1a = __builtin_amdgcn_wmma_f32_16x16x32_bf16(false, A1, false, B1a, (short)0, z, false, false);
        v8f d1b = __builtin_amdgcn_wmma_f32_16x16x32_bf16(false, A1, false, B1b, (short)0, z, false, false);

        // bias + relu, transpose D layout -> [pixel][feature] in LDS
#pragma unroll
        for (int r = 0; r < 8; ++r) {
            const int pix = h * 8 + r;                           // D: M = h*8 + r, N = lane&15
            lds[pix * 33 + n]      = fmaxf(0.0f, d1a[r] + bz1a);
            lds[pix * 33 + 16 + n] = fmaxf(0.0f, d1b[r] + bz1b);
        }
        asm volatile("s_wait_dscnt 0" ::: "memory");

        // rebuild A layout for layer 2
        v16bf A2;
#pragma unroll
        for (int e = 0; e < 8; ++e) {
            A2[e]     = (__bf16)lds[m * 33 + h * 8 + e];         // K = h*8 + e      (0..7 | 8..15)
            A2[8 + e] = (__bf16)lds[m * 33 + 16 + h * 8 + e];    // K = 16 + h*8 + e (16..23 | 24..31)
        }

        // ---- layer 2: 32 -> 32
        v8f d2a = __builtin_amdgcn_wmma_f32_16x16x32_bf16(false, A2, false, B2a, (short)0, z, false, false);
        v8f d2b = __builtin_amdgcn_wmma_f32_16x16x32_bf16(false, A2, false, B2b, (short)0, z, false, false);

#pragma unroll
        for (int r = 0; r < 8; ++r) {
            const int pix = h * 8 + r;
            lds[pix * 33 + n]      = fmaxf(0.0f, d2a[r] + bz2a);
            lds[pix * 33 + 16 + n] = fmaxf(0.0f, d2b[r] + bz2b);
        }
        asm volatile("s_wait_dscnt 0" ::: "memory");

        v16bf A3;
#pragma unroll
        for (int e = 0; e < 8; ++e) {
            A3[e]     = (__bf16)lds[m * 33 + h * 8 + e];
            A3[8 + e] = (__bf16)lds[m * 33 + 16 + h * 8 + e];
        }

        // ---- layer 3: 32 -> 8 (B zero-padded to 16 outputs)
        v8f d3 = __builtin_amdgcn_wmma_f32_16x16x32_bf16(false, A3, false, B3f, (short)0, z, false, false);

        // scatter raw dx into LDS [pixel][channel] (stride 9) for coalesced epilogue
        if (n < 8) {
#pragma unroll
            for (int r = 0; r < 8; ++r)
                lds[(h * 8 + r) * 9 + n] = d3[r] + bz3;
        }
        asm volatile("s_wait_dscnt 0" ::: "memory");

        // ---- epilogue: growth-rate scale, ch3 boost, clamp, store (16 pixels x 8 channels = 128 vals)
#pragma unroll
        for (int i = 0; i < 4; ++i) {
            const int lin = i * 32 + lane;
            const int c   = lin >> 4;
            const int mm  = lin & 15;
            const int xcol = tx * 16 + mm;
            float dxv = lds[mm * 9 + c] * g;
            const int gidx = (b * C_ + c) * HW_ + y * W_ + xcol;
            if (c == 3 && applyBoost) {
                // neighbor_alive > 0.01  <=>  any in-bounds 3x3 neighbor alive (1/9 > 0.01)
                const float* pa = xb + 3 * HW_;
                bool any = false;
#pragma unroll
                for (int dy = -1; dy <= 1; ++dy)
#pragma unroll
                    for (int dxo = -1; dxo <= 1; ++dxo)
                        any = any || (ldz(pa, y + dy, xcol + dxo) > 0.1f);
                if (any) dxv += 0.2f;
            }
            float nv = fminf(1.0f, fmaxf(-1.0f, x[gidx] + dxv));
            if (c == 3) nv = fmaxf(0.0f, nv);   // clip(clip(v,-1,1),0,1) == clip(v,0,1)
            out[gidx] = nv;
        }
    }
}

extern "C" void kernel_launch(void* const* d_in, const int* in_sizes, int n_in,
                              void* d_out, int out_size, void* d_ws, size_t ws_size,
                              hipStream_t stream) {
    const float* x  = (const float*)d_in[0];
    const float* w1 = (const float*)d_in[1];
    const float* b1 = (const float*)d_in[2];
    const float* w2 = (const float*)d_in[3];
    const float* b2 = (const float*)d_in[4];
    const float* w3 = (const float*)d_in[5];
    const float* b3 = (const float*)d_in[6];
    const float* gr = (const float*)d_in[7];
    unsigned* cnt = (unsigned*)d_ws;

    hipMemsetAsync(d_ws, 0, sizeof(unsigned), stream);   // ws is poisoned; zero the counter each call
    alive_count_kernel<<<2048, 256, 0, stream>>>(x, cnt);
    nca_fused_kernel<<<(B_ * H_) / 8, 256, 0, stream>>>(x, w1, b1, w2, b2, w3, b3, gr, cnt,
                                                        (float*)d_out);
}
